// StandardMixerBlock_26869315403871
// MI455X (gfx1250) — matrix operand order repaired and checked by reading, not compiled, not run on hardware
//
#include <hip/hip_runtime.h>
#include <hip/hip_bf16.h>

#define B_   256
#define P_   256
#define D_   514
#define H_   8
#define HD_  64
#define SH_  256
#define HH_  256
#define DFF_ 2048
#define LD_  512
#define LH_  512
#define LDA_ 40   // padded LDS row stride in halves (32 data + 8 pad = 80B rows, 16B aligned)

typedef __attribute__((ext_vector_type(16))) _Float16 v16h;
typedef __attribute__((ext_vector_type(8)))  float    v8f;
typedef __attribute__((ext_vector_type(4)))  int      v4i_;

__device__ __forceinline__ float gelu_f(float v) {
  return 0.5f * v * (1.0f + erff(v * 0.70710678118654752f));
}

// ------------- gfx1250 async global->LDS path (guarded) -------------
#if defined(__HIP_DEVICE_COMPILE__) && defined(__has_builtin)
#if __has_builtin(__builtin_amdgcn_global_load_async_to_lds_b128) && \
    __has_builtin(__builtin_amdgcn_s_wait_asynccnt)
#define ASYNC_LDS 1
#endif
#endif
#ifndef ASYNC_LDS
#define ASYNC_LDS 0
#endif

#if ASYNC_LDS
// builtin signature (from hipcc diagnostic): (int4 AS1*, int4 AS3*, Imm int, Imm int)
typedef __attribute__((address_space(1))) v4i_ g1v4;
typedef __attribute__((address_space(3))) v4i_ l3v4;
// generic global ptr value == AS1 value; generic LDS ptr low 32 bits == LDS offset
__device__ __forceinline__ g1v4* as_g1(const void* p) {
  return (g1v4*)(unsigned long long)p;
}
__device__ __forceinline__ l3v4* as_l3(void* p) {
  return (l3v4*)(unsigned)(unsigned long long)p;
}
#endif

__device__ __forceinline__ void lds_wait_async() {
#if ASYNC_LDS
  __builtin_amdgcn_s_wait_asynccnt(0);
#endif
}

// ---- WMMA fragment helpers (gfx1250 wave32 layouts, CDNA5 ISA) ----

// A 16x32 f16: lane m=lane&15; lanes0-15 hold K0..15, lanes16-31 hold K16..31
// (same k order as frag_b).
__device__ __forceinline__ v16h frag_a(const _Float16* As, int mBase, int lane) {
  const _Float16* p = As + (mBase + (lane & 15)) * LDA_ + ((lane >> 4) << 4);
  v16h a;
#pragma unroll
  for (int i = 0; i < 16; i++) a[i] = p[i];
  return a;
}

// B 32x16 f16 (stored transposed in LDS as [N][K]): lane n=lane&15;
// lanes0-15 hold K0..15, lanes16-31 hold K16..31.
__device__ __forceinline__ v16h frag_b(const _Float16* Bs, int nBase, int lane) {
  const _Float16* p = Bs + (nBase + (lane & 15)) * LDA_ + ((lane >> 4) << 4);
  v16h b;
#pragma unroll
  for (int i = 0; i < 16; i++) b[i] = p[i];
  return b;
}

__device__ __forceinline__ v8f wmma16(v16h a, v16h b, v8f c) {
  return __builtin_amdgcn_wmma_f32_16x16x32_f16(false, a, false, b, (short)0, c,
                                                false, false);
}

// ---- cooperative tile loaders: 64x32 tiles into LDS [64][LDA_] f16 ----

// f16 source, rows contiguous & 16B aligned: async DMA to LDS (128 thr * 2 * 16B)
__device__ __forceinline__ void ld_rows_f16t(_Float16* dst, const _Float16* src,
                                             int rs, int tid) {
#if ASYNC_LDS
  const int r = tid >> 2, cb = (tid & 3) << 3;   // cb in halves (8 halves = 16B)
#pragma unroll
  for (int i = 0; i < 2; i++) {
    const int rr = r + i * 32;
    __builtin_amdgcn_global_load_async_to_lds_b128(
        as_g1(src + (size_t)rr * rs + cb), as_l3(dst + rr * LDA_ + cb), 0, 0);
  }
#else
  for (int e = tid; e < 2048; e += 128) {
    int r = e >> 5, c = e & 31;
    dst[r * LDA_ + c] = src[(size_t)r * rs + c];
  }
#endif
}
// f32 source, rows contiguous in K: convert on the fly
__device__ __forceinline__ void ld_rows_f32(_Float16* dst, const float* src,
                                            int rs, int tid) {
  for (int e = tid; e < 2048; e += 128) {
    int r = e >> 5, c = e & 31;
    dst[r * LDA_ + c] = (_Float16)src[(size_t)r * rs + c];
  }
}
// memory contiguous along M (token-mix A): src[k*rs + r]
__device__ __forceinline__ void ld_cols_f32(_Float16* dst, const float* src,
                                            int rs, int tid) {
  for (int e = tid; e < 2048; e += 128) {
    int r = e & 63, c = e >> 6;
    dst[r * LDA_ + c] = (_Float16)src[(size_t)c * rs + r];
  }
}

// ============== weight pre-conversion kernels (run once per launch) ==============
__global__ void __launch_bounds__(256)
cvt_kernel(const float* __restrict__ src, _Float16* __restrict__ dst, int n) {
  int i = blockIdx.x * 256 + threadIdx.x;
  if (i < n) dst[i] = (_Float16)src[i];
}
// FW1 (2048 x 514) -> packed f16 (2048 x 512), dropping the 2 shortcut columns
__global__ void __launch_bounds__(256)
cvt_fw1_kernel(const float* __restrict__ FW1, _Float16* __restrict__ dst) {
  int i = blockIdx.x * 256 + threadIdx.x;   // over 2048*512
  int f = i >> 9, c = i & 511;
  dst[i] = (_Float16)FW1[(size_t)f * D_ + c];
}

// ================= kernel 0: FiLM modulation (tiny, VALU) =================
__global__ void __launch_bounds__(256)
mod_kernel(const float* __restrict__ x, const float* __restrict__ LW1,
           const float* __restrict__ Lb1, const float* __restrict__ LW2,
           const float* __restrict__ Lb2, float* __restrict__ gb) {
  __shared__ float hid[LH_];
  const int b = blockIdx.x;
  const float s0 = x[(size_t)b * P_ * D_ + 128 * D_ + 512];
  const float s1 = x[(size_t)b * P_ * D_ + 128 * D_ + 513];
  for (int j = threadIdx.x; j < LH_; j += 256)
    hid[j] = gelu_f(s0 * LW1[j * 2] + s1 * LW1[j * 2 + 1] + Lb1[j]);
  __syncthreads();
  for (int o = threadIdx.x; o < 2 * SH_; o += 256) {
    float acc = Lb2[o];
    const float* wrow = LW2 + (size_t)o * LH_;
    for (int j = 0; j < LH_; j++) acc += wrow[j] * hid[j];
    gb[b * 512 + o] = acc;   // [0:256) gamma, [256:512) beta
  }
}

// ======= kernel 1: token mix  shared[b][c][o] = (x[b]^T @ Ws^T)*g+b =======
__global__ void __launch_bounds__(128)
tokenmix_kernel(const float* __restrict__ x, const _Float16* __restrict__ Ws16,
                const float* __restrict__ bs, const float* __restrict__ gb,
                _Float16* __restrict__ sh) {
  __shared__ __align__(16) _Float16 As[64 * LDA_], Bs[64 * LDA_];
  const int tid = threadIdx.x, lane = tid & 31, wave = tid >> 5;
  const int waveM = (wave & 1) * 32, waveN = (wave >> 1) * 32;
  const int b = blockIdx.z, m0 = blockIdx.x * 64, n0 = blockIdx.y * 64;
  const float* xb = x + (size_t)b * P_ * D_;
  v8f acc[2][2] = {};
  for (int k0 = 0; k0 < P_; k0 += 32) {
    __syncthreads();
    ld_cols_f32(As, xb + (size_t)k0 * D_ + m0, D_, tid);      // A[c][p]
    ld_rows_f16t(Bs, Ws16 + (size_t)n0 * P_ + k0, P_, tid);   // Bt[o][p] = Ws row
    lds_wait_async();
    __syncthreads();
    v16h a0 = frag_a(As, waveM, lane), a1 = frag_a(As, waveM + 16, lane);
    v16h b0 = frag_b(Bs, waveN, lane), b1 = frag_b(Bs, waveN + 16, lane);
    acc[0][0] = wmma16(a0, b0, acc[0][0]);
    acc[0][1] = wmma16(a0, b1, acc[0][1]);
    acc[1][0] = wmma16(a1, b0, acc[1][0]);
    acc[1][1] = wmma16(a1, b1, acc[1][1]);
  }
  const int hi = lane >> 4, nl = lane & 15;
#pragma unroll
  for (int fm = 0; fm < 2; fm++)
#pragma unroll
    for (int fn = 0; fn < 2; fn++) {
      int n = n0 + waveN + fn * 16 + nl;
      float bias = bs[n], g = gb[b * 512 + n], be = gb[b * 512 + 256 + n];
#pragma unroll
      for (int j = 0; j < 8; j++) {
        int m = m0 + waveM + fm * 16 + j + hi * 8;
        sh[((size_t)b * 512 + m) * 256 + n] =
            (_Float16)((acc[fm][fn][j] + bias) * g + be);
      }
    }
}

// ========== kernel 2: per-head MLP layer 1, h = gelu(shared @ HW1^T) ==========
__global__ void __launch_bounds__(128)
head1_kernel(const _Float16* __restrict__ sh, const _Float16* __restrict__ HW1h,
             const float* __restrict__ Hb1, _Float16* __restrict__ hb) {
  __shared__ __align__(16) _Float16 As[64 * LDA_], Bs[64 * LDA_];
  const int tid = threadIdx.x, lane = tid & 31, wave = tid >> 5;
  const int waveM = (wave & 1) * 32, waveN = (wave >> 1) * 32;
  const int bh = blockIdx.y, h = bh & 7, n0 = blockIdx.x * 64;
  const _Float16* A = sh + (size_t)bh * 64 * 256;
  const _Float16* Bt = HW1h + (size_t)h * HH_ * SH_;
  v8f acc[2][2] = {};
  for (int k0 = 0; k0 < SH_; k0 += 32) {
    __syncthreads();
    ld_rows_f16t(As, A + k0, 256, tid);
    ld_rows_f16t(Bs, Bt + (size_t)n0 * SH_ + k0, SH_, tid);
    lds_wait_async();
    __syncthreads();
    v16h a0 = frag_a(As, waveM, lane), a1 = frag_a(As, waveM + 16, lane);
    v16h b0 = frag_b(Bs, waveN, lane), b1 = frag_b(Bs, waveN + 16, lane);
    acc[0][0] = wmma16(a0, b0, acc[0][0]);
    acc[0][1] = wmma16(a0, b1, acc[0][1]);
    acc[1][0] = wmma16(a1, b0, acc[1][0]);
    acc[1][1] = wmma16(a1, b1, acc[1][1]);
  }
  const int hi = lane >> 4, nl = lane & 15;
#pragma unroll
  for (int fm = 0; fm < 2; fm++)
#pragma unroll
    for (int fn = 0; fn < 2; fn++) {
      int n = n0 + waveN + fn * 16 + nl;
      float bias = Hb1[h * HH_ + n];
#pragma unroll
      for (int j = 0; j < 8; j++) {
        int m = waveM + fm * 16 + j + hi * 8;
        hb[((size_t)bh * 64 + m) * 256 + n] = (_Float16)gelu_f(acc[fm][fn][j] + bias);
      }
    }
}

// ===== kernel 3: per-head MLP layer 2 + residual + transpose to y[b][p][c] =====
__global__ void __launch_bounds__(128)
head2_kernel(const _Float16* __restrict__ hb, const _Float16* __restrict__ HW2h,
             const float* __restrict__ Hb2, const float* __restrict__ x,
             float* __restrict__ y) {
  __shared__ __align__(16) _Float16 As[64 * LDA_], Bs[64 * LDA_];
  const int tid = threadIdx.x, lane = tid & 31, wave = tid >> 5;
  const int waveM = (wave & 1) * 32, waveN = (wave >> 1) * 32;
  const int bh = blockIdx.y, b = bh >> 3, h = bh & 7, n0 = blockIdx.x * 64;
  const _Float16* A = hb + (size_t)bh * 64 * 256;
  const _Float16* Bt = HW2h + (size_t)h * P_ * HH_;
  v8f acc[2][2] = {};
  for (int k0 = 0; k0 < HH_; k0 += 32) {
    __syncthreads();
    ld_rows_f16t(As, A + k0, 256, tid);
    ld_rows_f16t(Bs, Bt + (size_t)n0 * HH_ + k0, HH_, tid);
    lds_wait_async();
    __syncthreads();
    v16h a0 = frag_a(As, waveM, lane), a1 = frag_a(As, waveM + 16, lane);
    v16h b0 = frag_b(Bs, waveN, lane), b1 = frag_b(Bs, waveN + 16, lane);
    acc[0][0] = wmma16(a0, b0, acc[0][0]);
    acc[0][1] = wmma16(a0, b1, acc[0][1]);
    acc[1][0] = wmma16(a1, b0, acc[1][0]);
    acc[1][1] = wmma16(a1, b1, acc[1][1]);
  }
  const int hi = lane >> 4, nl = lane & 15;
#pragma unroll
  for (int fm = 0; fm < 2; fm++)
#pragma unroll
    for (int fn = 0; fn < 2; fn++) {
      int n = n0 + waveN + fn * 16 + nl;                   // p index
      float bias = Hb2[h * P_ + n];
#pragma unroll
      for (int j = 0; j < 8; j++) {
        int m = waveM + fm * 16 + j + hi * 8;              // d index
        int c = h * 64 + m;
        float v = acc[fm][fn][j] + bias + x[(size_t)b * P_ * D_ + (size_t)n * D_ + c];
        y[((size_t)b * P_ + n) * 512 + c] = v;
      }
    }
}

// ============ kernel 4: FFN layer 1, u = gelu(x_full @ FW1^T + Fb1) ============
__global__ void __launch_bounds__(128)
ffn1_kernel(const float* __restrict__ x, const float* __restrict__ y,
            const _Float16* __restrict__ FW1h, const float* __restrict__ FW1,
            const float* __restrict__ Fb1, _Float16* __restrict__ ub) {
  __shared__ __align__(16) _Float16 As[64 * LDA_], Bs[64 * LDA_];
  const int tid = threadIdx.x, lane = tid & 31, wave = tid >> 5;
  const int waveM = (wave & 1) * 32, waveN = (wave >> 1) * 32;
  const int m0 = blockIdx.y * 64, n0 = blockIdx.x * 64;
  v8f acc[2][2] = {};
  for (int k0 = 0; k0 < 512; k0 += 32) {
    __syncthreads();
    ld_rows_f32(As, y + (size_t)m0 * 512 + k0, 512, tid);
    ld_rows_f16t(Bs, FW1h + (size_t)n0 * 512 + k0, 512, tid);
    lds_wait_async();
    __syncthreads();
    v16h a0 = frag_a(As, waveM, lane), a1 = frag_a(As, waveM + 16, lane);
    v16h b0 = frag_b(Bs, waveN, lane), b1 = frag_b(Bs, waveN + 16, lane);
    acc[0][0] = wmma16(a0, b0, acc[0][0]);
    acc[0][1] = wmma16(a0, b1, acc[0][1]);
    acc[1][0] = wmma16(a1, b0, acc[1][0]);
    acc[1][1] = wmma16(a1, b1, acc[1][1]);
  }
  const int hi = lane >> 4, nl = lane & 15;
#pragma unroll
  for (int fm = 0; fm < 2; fm++)
#pragma unroll
    for (int fn = 0; fn < 2; fn++) {
      int n = n0 + waveN + fn * 16 + nl;
      float w0 = FW1[(size_t)n * D_ + 512], w1 = FW1[(size_t)n * D_ + 513];
      float bias = Fb1[n];
#pragma unroll
      for (int j = 0; j < 8; j++) {
        int m = m0 + waveM + fm * 16 + j + hi * 8;         // flattened (b,p)
        float s0 = x[(size_t)m * D_ + 512], s1 = x[(size_t)m * D_ + 513];
        float v = acc[fm][fn][j] + bias + s0 * w0 + s1 * w1;
        ub[(size_t)m * DFF_ + n] = (_Float16)gelu_f(v);
      }
    }
}

// ======== kernel 5: FFN layer 2 + residual, out[..c] = u @ FW2^T + Fb2 + y ========
__global__ void __launch_bounds__(128)
ffn2_kernel(const _Float16* __restrict__ ub, const _Float16* __restrict__ FW2h,
            const float* __restrict__ Fb2, const float* __restrict__ y,
            float* __restrict__ out) {
  __shared__ __align__(16) _Float16 As[64 * LDA_], Bs[64 * LDA_];
  const int tid = threadIdx.x, lane = tid & 31, wave = tid >> 5;
  const int waveM = (wave & 1) * 32, waveN = (wave >> 1) * 32;
  const int m0 = blockIdx.y * 64, n0 = blockIdx.x * 64;
  v8f acc[2][2] = {};
  for (int k0 = 0; k0 < DFF_; k0 += 32) {
    __syncthreads();
    ld_rows_f16t(As, ub + (size_t)m0 * DFF_ + k0, DFF_, tid);
    ld_rows_f16t(Bs, FW2h + (size_t)n0 * DFF_ + k0, DFF_, tid);
    lds_wait_async();
    __syncthreads();
    v16h a0 = frag_a(As, waveM, lane), a1 = frag_a(As, waveM + 16, lane);
    v16h b0 = frag_b(Bs, waveN, lane), b1 = frag_b(Bs, waveN + 16, lane);
    acc[0][0] = wmma16(a0, b0, acc[0][0]);
    acc[0][1] = wmma16(a0, b1, acc[0][1]);
    acc[1][0] = wmma16(a1, b0, acc[1][0]);
    acc[1][1] = wmma16(a1, b1, acc[1][1]);
  }
  const int hi = lane >> 4, nl = lane & 15;
#pragma unroll
  for (int fm = 0; fm < 2; fm++)
#pragma unroll
    for (int fn = 0; fn < 2; fn++) {
      int n = n0 + waveN + fn * 16 + nl;                   // c index
      float bias = Fb2[n];
#pragma unroll
      for (int j = 0; j < 8; j++) {
        int m = m0 + waveM + fm * 16 + j + hi * 8;
        float v = acc[fm][fn][j] + bias + y[(size_t)m * 512 + n];
        out[(size_t)m * D_ + n] = v;
      }
    }
}

// ============ kernel 6: copy shortcut channels into the output ============
__global__ void __launch_bounds__(256)
sc_kernel(const float* __restrict__ x, float* __restrict__ out) {
  size_t m = (size_t)blockIdx.x * 256 + threadIdx.x;   // 0..B*P-1
  out[m * D_ + 512] = x[m * D_ + 512];
  out[m * D_ + 513] = x[m * D_ + 513];
}

extern "C" void kernel_launch(void* const* d_in, const int* in_sizes, int n_in,
                              void* d_out, int out_size, void* d_ws, size_t ws_size,
                              hipStream_t stream) {
  const float* x   = (const float*)d_in[0];
  const float* Ws  = (const float*)d_in[1];
  const float* bs  = (const float*)d_in[2];
  const float* LW1 = (const float*)d_in[3];
  const float* Lb1 = (const float*)d_in[4];
  const float* LW2 = (const float*)d_in[5];
  const float* Lb2 = (const float*)d_in[6];
  const float* HW1 = (const float*)d_in[7];
  const float* Hb1 = (const float*)d_in[8];
  const float* HW2 = (const float*)d_in[9];
  const float* Hb2 = (const float*)d_in[10];
  const float* FW1 = (const float*)d_in[11];
  const float* Fb1 = (const float*)d_in[12];
  const float* FW2 = (const float*)d_in[13];
  const float* Fb2 = (const float*)d_in[14];
  float* out = (float*)d_out;

  char* w = (char*)d_ws;
  const size_t MB = (size_t)1 << 20;
  float*    gb   = (float*)w;                       // 0.5 MB
  _Float16* Ws16 = (_Float16*)(w + 1 * MB);         // 128 KB
  _Float16* HW1h = (_Float16*)(w + 2 * MB);         // 1 MB
  _Float16* HW2h = (_Float16*)(w + 3 * MB);         // 1 MB
  _Float16* FW1h = (_Float16*)(w + 4 * MB);         // 2 MB (packed 2048x512)
  _Float16* FW2h = (_Float16*)(w + 6 * MB);         // 2 MB
  _Float16* sh   = (_Float16*)(w + 8 * MB);         // 64 MB f16
  _Float16* hb   = (_Float16*)(w + 72 * MB);        // 64 MB f16
  float*    yb   = (float*)   (w + 136 * MB);       // 128 MB f32
  _Float16* ub   = (_Float16*)(w + 264 * MB);       // 256 MB f16

  // one-shot weight conversion to f16 (deterministic, ~12 MB total)
  cvt_kernel<<<(SH_ * P_) / 256, 256, 0, stream>>>(Ws, Ws16, SH_ * P_);
  cvt_kernel<<<(H_ * HH_ * SH_) / 256, 256, 0, stream>>>(HW1, HW1h, H_ * HH_ * SH_);
  cvt_kernel<<<(H_ * P_ * HH_) / 256, 256, 0, stream>>>(HW2, HW2h, H_ * P_ * HH_);
  cvt_kernel<<<(LD_ * DFF_) / 256, 256, 0, stream>>>(FW2, FW2h, LD_ * DFF_);
  cvt_fw1_kernel<<<(DFF_ * 512) / 256, 256, 0, stream>>>(FW1, FW1h);

  mod_kernel<<<B_, 256, 0, stream>>>(x, LW1, Lb1, LW2, Lb2, gb);
  tokenmix_kernel<<<dim3(8, 4, B_), 128, 0, stream>>>(x, Ws16, bs, gb, sh);
  head1_kernel<<<dim3(4, B_ * H_), 128, 0, stream>>>(sh, HW1h, Hb1, hb);
  head2_kernel<<<dim3(4, B_ * H_), 128, 0, stream>>>(hb, HW2h, Hb2, x, yb);
  ffn1_kernel<<<dim3(DFF_ / 64, (B_ * P_) / 64), 128, 0, stream>>>(x, yb, FW1h, FW1, Fb1, ub);
  ffn2_kernel<<<dim3(LD_ / 64, (B_ * P_) / 64), 128, 0, stream>>>(ub, FW2h, Fb2, yb, out);
  sc_kernel<<<(B_ * P_) / 256, 256, 0, stream>>>(x, out);
}
